// ChannelMerger_37185826848894
// MI455X (gfx1250) — matrix operand order repaired and checked by reading, not compiled, not run on hardware
//
#include <hip/hip_runtime.h>
#include <math.h>

// ---------------------------------------------------------------------------
// ChannelMerger for MI455X (gfx1250, wave32, WMMA).
//
// out[b,o,t] = sum_c softmax_c(emb(layout) @ heads^T)[o,c] * x[b,c,t]
//   B=64, C=270, T=2000, O=270, D=2048
//
// Heavy part: W[270x270] x X[b][270x2000]  (18.7 GFLOP, 276 MB HBM traffic,
// roofline floor ~12us @ 23.3 TB/s).  bf16x3 split-precision WMMA
// (v_wmma_f32_16x16x32_bf16, f32 accum): acc += Whi*Xhi + Whi*Xlo + Wlo*Xhi.
//
// Loop structure: one wave owns a (b, 16-wide t-tile) slab and ALL 17
// o-tiles, so each x element is loaded + bf16-split exactly once.
// ---------------------------------------------------------------------------

typedef __attribute__((ext_vector_type(16))) __bf16         v16bf;
typedef __attribute__((ext_vector_type(16))) unsigned short v16us;
typedef __attribute__((ext_vector_type(8)))  float          v8f;
typedef __attribute__((ext_vector_type(4)))  unsigned int   v4u;

#define C_DIM 270
#define T_DIM 2000
#define B_DIM 64
#define O_DIM 270
#define D_DIM 2048
#define K_PAD 288   // 270 padded to 9*32 for the WMMA K loop
#define O_PAD 272   // 270 padded to 17*16
#define N_OT  17    // o-tiles per wave

static __device__ __forceinline__ unsigned short f32_to_bf16_rne(float f) {
    unsigned int u = __float_as_uint(f);
    unsigned int lsb = (u >> 16) & 1u;
    u += 0x7fffu + lsb;                 // round-to-nearest-even (finite inputs)
    return (unsigned short)(u >> 16);
}
static __device__ __forceinline__ float bf16_to_f32(unsigned short h) {
    return __uint_as_float(((unsigned int)h) << 16);
}

// ---------------------------------------------------------------------------
// Kernel 1: Fourier embedding, stored transposed embT[d][c] so the scores
// kernel reads it coalesced. One block per channel c. (Tiny: 0.5M sincos.)
// ---------------------------------------------------------------------------
__global__ void cm_embed_kernel(const float* __restrict__ layout,
                                float* __restrict__ embT) {
    const int c = blockIdx.x;                // 0..269
    const float scale = (float)(2.0 * M_PI / 1.4);   // 2*pi / (1 + 2*margin)
    const float px = layout[2 * c + 0] + 0.2f;
    const float py = layout[2 * c + 1] + 0.2f;
    for (int idx = threadIdx.x; idx < 1024; idx += blockDim.x) {
        const int i = idx >> 5;              // freq index x (0..31)
        const int j = idx & 31;              // freq index y (0..31)
        float loc = scale * (px * (float)i + py * (float)j);
        float s, cv;
        sincosf(loc, &s, &cv);
        embT[(size_t)idx * C_DIM + c]          = cv;  // d = idx        (cos)
        embT[(size_t)(1024 + idx) * C_DIM + c] = s;   // d = 1024+idx   (sin)
    }
}

// ---------------------------------------------------------------------------
// Kernel 2: scores row o = emb @ heads[o], invalid mask, softmax over c,
// emit bf16-split weights Whi/Wlo padded to [O_PAD][K_PAD] (zeros in pad so
// the GEMM needs no M/K edge guards on the A side). One block per row.
// ---------------------------------------------------------------------------
__global__ void cm_scores_kernel(const float* __restrict__ embT,
                                 const float* __restrict__ heads,
                                 const float* __restrict__ layout,
                                 unsigned short* __restrict__ Whi,
                                 unsigned short* __restrict__ Wlo) {
    const int o = blockIdx.x;                // 0..271
    if (o >= O_DIM) {                        // zero pad rows 270, 271
        for (int k = threadIdx.x; k < K_PAD; k += blockDim.x) {
            Whi[(size_t)o * K_PAD + k] = 0;
            Wlo[(size_t)o * K_PAD + k] = 0;
        }
        return;
    }

    __shared__ float hrow[D_DIM];
    __shared__ float sc[C_DIM];
    for (int d = threadIdx.x; d < D_DIM; d += blockDim.x)
        hrow[d] = heads[(size_t)o * D_DIM + d];
    __syncthreads();

    for (int c = threadIdx.x; c < C_DIM; c += blockDim.x) {
        float acc = 0.f;
        // embT is [D][C]: lanes read consecutive c -> coalesced, L2-resident.
        for (int d = 0; d < D_DIM; ++d)
            acc = fmaf(embT[(size_t)d * C_DIM + c], hrow[d], acc);
        const bool invalid =
            (layout[2 * c] == -0.1f) && (layout[2 * c + 1] == -0.1f);
        sc[c] = invalid ? -INFINITY : acc;
    }
    __syncthreads();

    // Redundant per-thread softmax stats over 270 LDS values (cheap).
    float m = -INFINITY;
    for (int c = 0; c < C_DIM; ++c) m = fmaxf(m, sc[c]);
    float sum = 0.f;
    for (int c = 0; c < C_DIM; ++c) sum += expf(sc[c] - m);
    const float inv = 1.0f / sum;

    for (int k = threadIdx.x; k < K_PAD; k += blockDim.x) {
        float w = (k < C_DIM) ? expf(sc[k] - m) * inv : 0.f;
        unsigned short hi = f32_to_bf16_rne(w);
        float lof = w - bf16_to_f32(hi);
        Whi[(size_t)o * K_PAD + k] = hi;
        Wlo[(size_t)o * K_PAD + k] = f32_to_bf16_rne(lof);
    }
}

// ---------------------------------------------------------------------------
// Kernel 3: out[b] = W x[b].  One wave per (b, 16-wide t-tile); the wave
// carries 17 accumulator tiles (all o). Per K-step the x hi/lo B-fragments
// are built ONCE and reused by 17 x 3 WMMAs.
//
// A fragment (16-bit 16x32 layout, per ISA): lane L holds row M=L&15;
//   lanes 0-15: K {0..7} in v0-3, {16..23} in v4-7
//   lanes 16-31: K {8..15} in v0-3, {24..31} in v4-7
//   -> two contiguous 16B chunks per lane from row-major W[M][K_PAD].
// B fragment (32x16): element i of v16bf is B[(lane>=16?8:0)+(i<8?i:i+8)][n],
//   built from 16 coalesced f32 row-reads of x + register split.
// ---------------------------------------------------------------------------
__global__ __launch_bounds__(32)
void cm_gemm_kernel(const float* __restrict__ x,
                    const unsigned short* __restrict__ Whi,
                    const unsigned short* __restrict__ Wlo,
                    float* __restrict__ out) {
    const int lane   = threadIdx.x;          // 0..31
    const int n      = lane & 15;            // column (t) within tile
    const bool hiH   = lane >= 16;
    const int t0     = blockIdx.x * 16;      // 0..1984
    const int b      = blockIdx.y;           // 0..63

    // Per-lane A base: row (n), half-chunk offset per documented lane layout.
    const size_t abase = (size_t)n * K_PAD + (hiH ? 8 : 0);
    const float* xcol  = x + ((size_t)b * C_DIM) * T_DIM + t0 + n;

    v8f acc[N_OT];
#pragma unroll
    for (int i = 0; i < N_OT; ++i) acc[i] = {};

    for (int k0 = 0; k0 < K_PAD; k0 += 32) {
        // --- Build B fragments (x hi/lo) once for this K-step -------------
        union { v16bf v; v16us u; } bhi, blo;
        const int kbase = k0 + (hiH ? 8 : 0);
#pragma unroll
        for (int i = 0; i < 16; ++i) {
            const int c  = kbase + (i < 8 ? i : i + 8);
            const int cc = c < (C_DIM - 1) ? c : (C_DIM - 1);
            float xv = xcol[(size_t)cc * T_DIM];     // 64B-coalesced row read
            xv = (c < C_DIM) ? xv : 0.f;             // K pad -> zero
            unsigned short h = f32_to_bf16_rne(xv);
            bhi.u[i] = h;
            blo.u[i] = f32_to_bf16_rne(xv - bf16_to_f32(h));
        }
        // Prefetch next K-chunk's x rows (global_prefetch_b8).
        if (k0 + 32 < K_PAD) {
            const int pc = (kbase + 32) < (C_DIM - 1) ? (kbase + 32) : (C_DIM - 1);
            __builtin_prefetch(xcol + (size_t)pc * T_DIM, 0, 1);
        }

        // --- 17 o-tiles reuse the B fragments -----------------------------
#pragma unroll
        for (int ot = 0; ot < N_OT; ++ot) {
            const unsigned short* wh =
                Whi + abase + (size_t)(ot * 16) * K_PAD + k0;
            const unsigned short* wl =
                Wlo + abase + (size_t)(ot * 16) * K_PAD + k0;
            union { v16bf v; v4u q[2]; } ahi, alo;
            ahi.q[0] = *(const v4u*)(wh);
            ahi.q[1] = *(const v4u*)(wh + 16);
            alo.q[0] = *(const v4u*)(wl);
            alo.q[1] = *(const v4u*)(wl + 16);

            // bf16x3 split product, f32 accumulate (drop lo*lo ~ 2^-32).
            acc[ot] = __builtin_amdgcn_wmma_f32_16x16x32_bf16(
                          false, ahi.v, false, bhi.v, (short)0, acc[ot],
                          false, false);
            acc[ot] = __builtin_amdgcn_wmma_f32_16x16x32_bf16(
                          false, ahi.v, false, blo.v, (short)0, acc[ot],
                          false, false);
            acc[ot] = __builtin_amdgcn_wmma_f32_16x16x32_bf16(
                          false, alo.v, false, bhi.v, (short)0, acc[ot],
                          false, false);
        }
    }

    // --- Store: C/D layout: VGPR j -> row M = j + (hiH ? 8 : 0) -----------
    float* obase = out + ((size_t)b * O_DIM) * T_DIM + t0 + n;
#pragma unroll
    for (int ot = 0; ot < N_OT; ++ot) {
#pragma unroll
        for (int j = 0; j < 8; ++j) {
            const int o = ot * 16 + j + (hiH ? 8 : 0);
            if (ot < N_OT - 1) {
                // o <= 255 < 270: guard folds away at compile time.
                obase[(size_t)o * T_DIM] = acc[ot][j];
            } else if (o < O_DIM) {
                obase[(size_t)o * T_DIM] = acc[ot][j];
            }
        }
    }
}

// ---------------------------------------------------------------------------
extern "C" void kernel_launch(void* const* d_in, const int* in_sizes, int n_in,
                              void* d_out, int out_size, void* d_ws, size_t ws_size,
                              hipStream_t stream) {
    const float* x      = (const float*)d_in[0];   // [64,270,2000]
    const float* layout = (const float*)d_in[1];   // [270,2]
    const float* heads  = (const float*)d_in[2];   // [270,2048]
    float* out          = (float*)d_out;           // [64,270,2000]

    char* ws = (char*)d_ws;
    float* embT = (float*)ws;                                  // [2048][270] f32
    size_t embBytes = (size_t)D_DIM * C_DIM * sizeof(float);   // ~2.2 MB
    unsigned short* Whi = (unsigned short*)(ws + embBytes);    // [272][288] bf16
    unsigned short* Wlo = Whi + (size_t)O_PAD * K_PAD;         // [272][288] bf16

    cm_embed_kernel<<<C_DIM, 256, 0, stream>>>(layout, embT);
    cm_scores_kernel<<<O_PAD, 256, 0, stream>>>(embT, heads, layout, Whi, Wlo);

    dim3 grid(T_DIM / 16, B_DIM);   // (125 t-tiles, 64 batches), 1 wave each
    cm_gemm_kernel<<<grid, 32, 0, stream>>>(x, Whi, Wlo, out);
}
